// FocalADMotionFeatureRefine_52802327937513
// MI455X (gfx1250) — matrix operand order repaired and checked by reading, not compiled, not run on hardware
//
#include <hip/hip_runtime.h>

#define B_DIM   256
#define N_NODES 512
#define E_EDGES 511
#define M_ROWS  (B_DIM * E_EDGES)   // 130816 = 128 * 1022

typedef __attribute__((ext_vector_type(16))) _Float16 v16h;
typedef __attribute__((ext_vector_type(8)))  _Float16 v8h;
typedef __attribute__((ext_vector_type(8)))  float    v8f;

// ---------------------------------------------------------------------------
// f16 WMMA GEMM:  C[M x N] = op(A[M x K(lda)] * Bw[K x N] + bias)
// Block tile 128x128, 8 waves in a 4(M) x 2(N) grid, each wave owns a 32x64
// sub-tile = 2x4 WMMA tiles -> 8 v_wmma_f32_16x16x32_f16 per wave per K-step.
// Requires: M%128==0, N%128==0, K%32==0, lda%8==0.
// ---------------------------------------------------------------------------
__global__ __launch_bounds__(256)
void gemm_f16_wmma(const _Float16* __restrict__ A, int lda,
                   const _Float16* __restrict__ Bw,      // K x N row-major
                   const float* __restrict__ bias,       // [N] or null
                   void* __restrict__ Cout, int c_is_f32,
                   const float* __restrict__ row_scale,  // [M] or null
                   int M, int N, int K, int relu)
{
    __shared__ __align__(16) _Float16 As[128][40];   // 128 rows x 32 k (+8 pad)
    __shared__ __align__(16) _Float16 Bt[128][40];   // transposed: [n][k]

    const int tid  = threadIdx.x;
    const int wave = tid >> 5;
    const int lane = tid & 31;
    const int l15  = lane & 15;
    const int g    = lane >> 4;
    const int bm = blockIdx.x * 128;
    const int bn = blockIdx.y * 128;
    const int wm = (wave & 3) * 32;     // 4 waves along M
    const int wn = (wave >> 2) * 64;    // 2 waves along N

    v8f acc[2][4];
#pragma unroll
    for (int j = 0; j < 4; ++j) {
        float bv = bias ? bias[bn + wn + j * 16 + l15] : 0.0f;
#pragma unroll
        for (int i = 0; i < 2; ++i)
#pragma unroll
            for (int t = 0; t < 8; ++t) acc[i][j][t] = bv;
    }

    // staging coordinates (constant over K loop)
    const int ar = tid >> 1;              // A row 0..127
    const int as = (tid & 1) * 16;        // A k-seg 0/16
    const int bk = tid >> 3;              // B k-row 0..31
    const int bs = (tid & 7) * 16;        // B n-seg 0..112

    for (int k0 = 0; k0 < K; k0 += 32) {
        __syncthreads();
        // stage A tile: 128 rows x 32 halves (2 x b128 per thread)
        {
            const _Float16* src = A + (size_t)(bm + ar) * lda + k0 + as;
            *(v8h*)&As[ar][as]     = *(const v8h*)src;
            *(v8h*)&As[ar][as + 8] = *(const v8h*)(src + 8);
            __builtin_prefetch(src + 32, 0, 1);   // next K slice
        }
        // stage B tile transposed: 32 k-rows x 128 n-cols
        {
            const _Float16* src = Bw + (size_t)(k0 + bk) * N + bn + bs;
            v8h b0 = *(const v8h*)src;
            v8h b1 = *(const v8h*)(src + 8);
            __builtin_prefetch(src + (size_t)32 * N, 0, 1);
#pragma unroll
            for (int j = 0; j < 8; ++j) Bt[bs + j][bk]     = b0[j];
#pragma unroll
            for (int j = 0; j < 8; ++j) Bt[bs + 8 + j][bk] = b1[j];
        }
        __syncthreads();

        // A fragments: lane l holds row wm + i*16 + (l&15); halves:
        //   h=0..7  -> K = g*8 + h ; h=8..15 -> K = 16 + g*8 + (h-8)
        v16h am[2];
#pragma unroll
        for (int i = 0; i < 2; ++i) {
            int mr = wm + i * 16 + l15;
            v8h alo = *(const v8h*)&As[mr][g * 8];
            v8h ahi = *(const v8h*)&As[mr][16 + g * 8];
            am[i] = __builtin_shufflevector(alo, ahi,
                0,1,2,3,4,5,6,7,8,9,10,11,12,13,14,15);
        }
        // B fragments: lane l holds col wn + j*16 + (l&15); halves h=0..7 -> K = g*8 + h ; h=8..15 -> K = 16 + g*8 + (h-8)
        v16h bf[4];
#pragma unroll
        for (int j = 0; j < 4; ++j) {
            int nc = wn + j * 16 + l15;
            v8h blo = *(const v8h*)&Bt[nc][g * 8];
            v8h bhi = *(const v8h*)&Bt[nc][16 + g * 8];
            bf[j] = __builtin_shufflevector(blo, bhi,
                0,1,2,3,4,5,6,7,8,9,10,11,12,13,14,15);
        }
#pragma unroll
        for (int i = 0; i < 2; ++i)
#pragma unroll
            for (int j = 0; j < 4; ++j)
                acc[i][j] = __builtin_amdgcn_wmma_f32_16x16x32_f16(
                    false, am[i], false, bf[j], (short)0, acc[i][j], false, false);
    }

    // epilogue: VGPR t -> M = i*16 + (lane<16 ? t : 8+t), N = wn + j*16 + (lane&15)
#pragma unroll
    for (int i = 0; i < 2; ++i) {
        int mbase = bm + wm + i * 16 + g * 8;
#pragma unroll
        for (int j = 0; j < 4; ++j) {
            int ncol = bn + wn + j * 16 + l15;
#pragma unroll
            for (int t = 0; t < 8; ++t) {
                int mr2 = mbase + t;
                float v = acc[i][j][t];
                if (relu && v < 0.0f) v = 0.0f;
                if (row_scale) v *= row_scale[mr2];
                if (c_is_f32) ((float*)Cout)[(size_t)mr2 * N + ncol] = v;
                else ((_Float16*)Cout)[(size_t)mr2 * N + ncol] = (_Float16)v;
            }
        }
    }
}

// ---------------------------------------------------------------------------
// Weight convert + pad: dst[Kp x N] f16; rows >= K zero. colOff for Ws2[:,1:].
// ---------------------------------------------------------------------------
__global__ __launch_bounds__(256)
void convert_w(const float* __restrict__ src, int srcStride, int colOff,
               _Float16* __restrict__ dst, int K, int Kp, int N)
{
    int i = blockIdx.x * 256 + threadIdx.x;
    if (i >= Kp * N) return;
    int k = i / N, n = i % N;
    float v = (k < K) ? src[(size_t)k * srcStride + colOff + n] : 0.0f;
    dst[i] = (_Float16)v;
}

// node_feat = node_raw @ Wn + bn  -> f16 [B*N, 64]
__global__ __launch_bounds__(256)
void node_feat_kernel(const float* __restrict__ node_raw,
                      const float* __restrict__ Wn, const float* __restrict__ bn,
                      _Float16* __restrict__ out)
{
    __shared__ float w[8 * 64];
    __shared__ float bb[64];
    int tid = threadIdx.x;
    if (tid < 64) bb[tid] = bn[tid];
    w[tid] = Wn[tid];
    w[tid + 256] = Wn[tid + 256];
    __syncthreads();
    int i = blockIdx.x * 256 + tid;           // B*N*64 elements, exact grid
    int row = i >> 6, d = i & 63;
    const float* nr = node_raw + (size_t)row * 8;
    float s = bb[d];
    for (int j = 0; j < 8; ++j) s += nr[j] * w[j * 64 + d];
    out[i] = (_Float16)s;
}

// edge_feat = edge_raw @ We + be -> f16 [B*E, 16]
__global__ __launch_bounds__(256)
void edge_feat_kernel(const float* __restrict__ edge_raw,
                      const float* __restrict__ We, const float* __restrict__ be,
                      _Float16* __restrict__ out, int total)
{
    int i = blockIdx.x * 256 + threadIdx.x;
    if (i >= total) return;
    int row = i >> 4, d = i & 15;
    const float* er = edge_raw + (size_t)row * 3;
    float s = be[d];
    for (int j = 0; j < 3; ++j) s += er[j] * We[j * 16 + d];
    out[i] = (_Float16)s;
}

__global__ __launch_bounds__(256)
void extract_ego_kernel(const _Float16* __restrict__ node16,
                        const int* __restrict__ ego_id,
                        _Float16* __restrict__ ego16)
{
    int i = blockIdx.x * 256 + threadIdx.x;   // B*64 elements
    int b = i >> 6, d = i & 63;
    int eid = *ego_id;
    ego16[i] = node16[((size_t)b * N_NODES + eid) * 64 + d];
}

// edge_input padded to 160: [ego(64) | edge(16) | neighbor(64) | 0(16)]
__global__ __launch_bounds__(256)
void build_edge_input_kernel(const _Float16* __restrict__ node16,
                             const _Float16* __restrict__ edge16,
                             const _Float16* __restrict__ ego16,
                             const int* __restrict__ ego_id,
                             _Float16* __restrict__ out)
{
    int i = blockIdx.x * 256 + threadIdx.x;   // M*160 elements, exact grid
    int r = i / 160, c = i - r * 160;
    int b = r / E_EDGES, e = r - b * E_EDGES;
    int eid = *ego_id;
    _Float16 v = (_Float16)0.0f;
    if (c < 64) {
        v = ego16[b * 64 + c];
    } else if (c < 80) {
        v = edge16[(size_t)r * 16 + (c - 64)];
    } else if (c < 144) {
        int nb = e + (e >= eid ? 1 : 0);
        v = node16[((size_t)b * N_NODES + nb) * 64 + (c - 80)];
    }
    out[i] = v;
}

// edge_input2 padded to 352: [neighbor(64) | edge(16) | c_ego(256) | 0(16)]
__global__ __launch_bounds__(256)
void build_edge_input2_kernel(const _Float16* __restrict__ node16,
                              const _Float16* __restrict__ edge16,
                              const _Float16* __restrict__ cego16,
                              const int* __restrict__ ego_id,
                              _Float16* __restrict__ out)
{
    int i = blockIdx.x * 256 + threadIdx.x;   // M*352 elements, exact grid
    int r = i / 352, c = i - r * 352;
    int b = r / E_EDGES, e = r - b * E_EDGES;
    int eid = *ego_id;
    _Float16 v = (_Float16)0.0f;
    if (c < 64) {
        int nb = e + (e >= eid ? 1 : 0);
        v = node16[((size_t)b * N_NODES + nb) * 64 + c];
    } else if (c < 80) {
        v = edge16[(size_t)r * 16 + (c - 64)];
    } else if (c < 336) {
        v = cego16[b * 256 + (c - 80)];
    }
    out[i] = v;
}

// h = [ego_feat | c_ego] -> [B, 320]
__global__ __launch_bounds__(256)
void build_h_kernel(const _Float16* __restrict__ ego16,
                    const _Float16* __restrict__ cego16,
                    _Float16* __restrict__ out)
{
    int i = blockIdx.x * 256 + threadIdx.x;   // B*320 elements
    int b = i / 320, c = i - b * 320;
    out[i] = (c < 64) ? ego16[b * 64 + c] : cego16[b * 256 + (c - 64)];
}

// Per-batch multi-head attention: scores, softmax, ctx. One block per b.
__global__ __launch_bounds__(256)
void attn_kernel(const float* __restrict__ qf,      // [B,256]
                 const _Float16* __restrict__ k16,  // [B*E,256]
                 const _Float16* __restrict__ v16,  // [B*E,256]
                 _Float16* __restrict__ ctx16)      // [B,256]
{
    __shared__ float sc[4][512];
    __shared__ float red[256];
    __shared__ float qs[256];
    int b = blockIdx.x, tid = threadIdx.x;
    qs[tid] = qf[b * 256 + tid];
    __syncthreads();

    for (int e = tid; e < 512; e += 256) {
        if (e < E_EDGES) {
            const _Float16* kr = k16 + ((size_t)b * E_EDGES + e) * 256;
            for (int h = 0; h < 4; ++h) {
                float s = 0.0f;
                for (int d = 0; d < 64; ++d)
                    s += qs[h * 64 + d] * (float)kr[h * 64 + d];
                sc[h][e] = s * 0.125f;
            }
        } else {
            for (int h = 0; h < 4; ++h) sc[h][e] = -1e30f;
        }
    }
    __syncthreads();

    for (int h = 0; h < 4; ++h) {
        red[tid] = fmaxf(sc[h][tid], sc[h][tid + 256]);
        __syncthreads();
        for (int s = 128; s > 0; s >>= 1) {
            if (tid < s) red[tid] = fmaxf(red[tid], red[tid + s]);
            __syncthreads();
        }
        float mx = red[0];
        __syncthreads();
        float e0 = __expf(sc[h][tid] - mx);
        float e1 = __expf(sc[h][tid + 256] - mx);
        sc[h][tid] = e0; sc[h][tid + 256] = e1;
        red[tid] = e0 + e1;
        __syncthreads();
        for (int s = 128; s > 0; s >>= 1) {
            if (tid < s) red[tid] += red[tid + s];
            __syncthreads();
        }
        float inv = 1.0f / red[0];
        __syncthreads();
        sc[h][tid] *= inv; sc[h][tid + 256] *= inv;
        __syncthreads();
    }

    int h = tid >> 6, d = tid & 63;
    float c = 0.0f;
    for (int e = 0; e < E_EDGES; ++e)
        c += sc[h][e] * (float)v16[((size_t)b * E_EDGES + e) * 256 + h * 64 + d];
    ctx16[b * 256 + tid] = (_Float16)c;
}

// score[r] = hidden[r,:] . Ws2[:,0] + bs2[0]
__global__ __launch_bounds__(256)
void score_kernel(const _Float16* __restrict__ hidden,
                  const float* __restrict__ Ws2, const float* __restrict__ bs2,
                  float* __restrict__ scores)
{
    __shared__ float w0[256];
    int tid = threadIdx.x;
    w0[tid] = Ws2[(size_t)tid * 1537];
    __syncthreads();
    int r = blockIdx.x * 256 + tid;           // M rows, exact grid (511 blocks)
    const _Float16* hr = hidden + (size_t)r * 256;
    float s = bs2[0];
    for (int j = 0; j < 256; ++j) s += (float)hr[j] * w0[j];
    scores[r] = s;
}

// Per-batch top-5 + alpha softmax. One block per b.
__global__ __launch_bounds__(256)
void topk_kernel(const float* __restrict__ scores,
                 float* __restrict__ out_idx, float* __restrict__ out_alpha,
                 float* __restrict__ alphabuf, int* __restrict__ idxbuf)
{
    __shared__ float sv[512];
    __shared__ float sr[256];
    __shared__ int   si[256];
    int b = blockIdx.x, tid = threadIdx.x;
    sv[tid]       = (tid       < E_EDGES) ? scores[b * E_EDGES + tid]       : -1e30f;
    sv[tid + 256] = (tid + 256 < E_EDGES) ? scores[b * E_EDGES + tid + 256] : -1e30f;
    __syncthreads();
    float vals[5]; int idxs[5];
    for (int kk = 0; kk < 5; ++kk) {
        float v0 = sv[tid], v1 = sv[tid + 256];
        float mv; int mi;
        if (v0 >= v1) { mv = v0; mi = tid; } else { mv = v1; mi = tid + 256; }
        sr[tid] = mv; si[tid] = mi;
        __syncthreads();
        for (int s = 128; s > 0; s >>= 1) {
            if (tid < s && sr[tid + s] > sr[tid]) { sr[tid] = sr[tid + s]; si[tid] = si[tid + s]; }
            __syncthreads();
        }
        vals[kk] = sr[0]; idxs[kk] = si[0];
        __syncthreads();
        if (tid == 0) sv[idxs[kk]] = -1e30f;
        __syncthreads();
    }
    if (tid == 0) {
        float mx = vals[0], s = 0.0f, ex[5];
        for (int kk = 0; kk < 5; ++kk) { ex[kk] = __expf(vals[kk] - mx); s += ex[kk]; }
        for (int kk = 0; kk < 5; ++kk) {
            float a = ex[kk] / s;
            out_idx[b * 5 + kk]   = (float)idxs[kk];
            out_alpha[b * 5 + kk] = a;
            alphabuf[b * 5 + kk]  = a;
            idxbuf[b * 5 + kk]    = idxs[kk];
        }
    }
}

// gather hidden rows of the selected edges -> [B*5, 256]
__global__ __launch_bounds__(256)
void gather_kernel(const _Float16* __restrict__ hidden,
                   const int* __restrict__ idxbuf,
                   _Float16* __restrict__ out)
{
    int i = blockIdx.x * 256 + threadIdx.x;   // 1280*256 elements, exact grid
    int r = i >> 8, c = i & 255;
    int b = r / 5;
    int e = idxbuf[r];
    out[i] = hidden[((size_t)b * E_EDGES + e) * 256 + c];
}

// ---------------------------------------------------------------------------
static inline char* carve(char*& p, size_t bytes) {
    char* r = p;
    p += (bytes + 255) & ~(size_t)255;
    return r;
}

extern "C" void kernel_launch(void* const* d_in, const int* in_sizes, int n_in,
                              void* d_out, int out_size, void* d_ws, size_t ws_size,
                              hipStream_t stream)
{
    (void)in_sizes; (void)n_in; (void)out_size;
    const float* node_raw = (const float*)d_in[0];
    const float* edge_raw = (const float*)d_in[1];
    const int*   ego_id   = (const int*)  d_in[2];
    const float* Wn  = (const float*)d_in[3];  const float* bn  = (const float*)d_in[4];
    const float* We  = (const float*)d_in[5];  const float* be  = (const float*)d_in[6];
    const float* Wq  = (const float*)d_in[7];  const float* bq  = (const float*)d_in[8];
    const float* Wkv = (const float*)d_in[9];  const float* bkv = (const float*)d_in[10];
    const float* Wiq = (const float*)d_in[11]; const float* biq = (const float*)d_in[12];
    const float* Wik = (const float*)d_in[13]; const float* bik = (const float*)d_in[14];
    const float* Wiv = (const float*)d_in[15]; const float* biv = (const float*)d_in[16];
    const float* Wo  = (const float*)d_in[17]; const float* bo  = (const float*)d_in[18];
    const float* Wf1 = (const float*)d_in[19]; const float* bf1 = (const float*)d_in[20];
    const float* Wf2 = (const float*)d_in[21]; const float* bf2 = (const float*)d_in[22];
    const float* Ws1 = (const float*)d_in[23]; const float* bs1 = (const float*)d_in[24];
    const float* Ws2 = (const float*)d_in[25]; const float* bs2 = (const float*)d_in[26];

    const int M = M_ROWS;                 // 130816

    // ---- carve workspace ----
    char* p = (char*)d_ws;
    _Float16* node16  = (_Float16*)carve(p, (size_t)B_DIM * N_NODES * 64 * 2);
    _Float16* edge16  = (_Float16*)carve(p, (size_t)M * 16 * 2);
    _Float16* ego16   = (_Float16*)carve(p, (size_t)B_DIM * 64 * 2);
    _Float16* ei1     = (_Float16*)carve(p, (size_t)M * 160 * 2);
    _Float16* kv16    = (_Float16*)carve(p, (size_t)M * 512 * 2);
    _Float16* k16     = (_Float16*)carve(p, (size_t)M * 256 * 2);
    _Float16* v16b    = (_Float16*)carve(p, (size_t)M * 256 * 2);
    _Float16* Q16     = (_Float16*)carve(p, (size_t)B_DIM * 256 * 2);
    float*    qf32    = (float*)   carve(p, (size_t)B_DIM * 256 * 4);
    _Float16* ctx16   = (_Float16*)carve(p, (size_t)B_DIM * 256 * 2);
    _Float16* cego16  = (_Float16*)carve(p, (size_t)B_DIM * 256 * 2);
    _Float16* hbuf    = (_Float16*)carve(p, (size_t)B_DIM * 320 * 2);
    _Float16* hmid16  = (_Float16*)carve(p, (size_t)B_DIM * 256 * 2);
    _Float16* ei2     = (_Float16*)carve(p, (size_t)M * 352 * 2);
    _Float16* hidden  = (_Float16*)carve(p, (size_t)M * 256 * 2);
    float*    scores  = (float*)   carve(p, (size_t)M * 4);
    float*    alphab  = (float*)   carve(p, 1280 * 4);
    int*      idxb    = (int*)     carve(p, 1280 * 4);
    _Float16* gath16  = (_Float16*)carve(p, (size_t)1280 * 256 * 2);
    _Float16* Wq16    = (_Float16*)carve(p, (size_t)64 * 256 * 2);
    _Float16* Wkv16   = (_Float16*)carve(p, (size_t)160 * 512 * 2);
    _Float16* Wiq16   = (_Float16*)carve(p, (size_t)256 * 256 * 2);
    _Float16* Wik16   = (_Float16*)carve(p, (size_t)256 * 256 * 2);
    _Float16* Wiv16   = (_Float16*)carve(p, (size_t)256 * 256 * 2);
    _Float16* Wo16    = (_Float16*)carve(p, (size_t)256 * 256 * 2);
    _Float16* Wf116   = (_Float16*)carve(p, (size_t)320 * 256 * 2);
    _Float16* Wf216   = (_Float16*)carve(p, (size_t)256 * 256 * 2);
    _Float16* Ws116   = (_Float16*)carve(p, (size_t)352 * 256 * 2);
    _Float16* Ws2c16  = (_Float16*)carve(p, (size_t)256 * 1536 * 2);
    if ((size_t)(p - (char*)d_ws) > ws_size) return;   // insufficient scratch

    float* out_h     = (float*)d_out;                      // [B,256]
    float* out_sel   = out_h + (size_t)B_DIM * 256;        // [B,5,6,256]
    float* out_idx   = out_sel + (size_t)B_DIM * 5 * 6 * 256;  // [B,5]
    float* out_alpha = out_idx + B_DIM * 5;                // [B,5]

    #define CVT(src, stride, off, dst, K, Kp, N) \
        convert_w<<<((Kp)*(N)+255)/256, 256, 0, stream>>>(src, stride, off, dst, K, Kp, N)
    CVT(Wq,  256,  0, Wq16,  64,  64,  256);
    CVT(Wkv, 512,  0, Wkv16, 144, 160, 512);
    CVT(Wiq, 256,  0, Wiq16, 256, 256, 256);
    CVT(Wik, 256,  0, Wik16, 256, 256, 256);
    CVT(Wiv, 256,  0, Wiv16, 256, 256, 256);
    CVT(Wo,  256,  0, Wo16,  256, 256, 256);
    CVT(Wf1, 256,  0, Wf116, 320, 320, 256);
    CVT(Wf2, 256,  0, Wf216, 256, 256, 256);
    CVT(Ws1, 256,  0, Ws116, 336, 352, 256);
    CVT(Ws2, 1537, 1, Ws2c16, 256, 256, 1536);
    #undef CVT

    // embeds
    node_feat_kernel<<<(B_DIM * N_NODES * 64) / 256, 256, 0, stream>>>(node_raw, Wn, bn, node16);
    edge_feat_kernel<<<(M * 16 + 255) / 256, 256, 0, stream>>>(edge_raw, We, be, edge16, M * 16);
    extract_ego_kernel<<<(B_DIM * 64) / 256, 256, 0, stream>>>(node16, ego_id, ego16);
    build_edge_input_kernel<<<(M * 160) / 256, 256, 0, stream>>>(node16, edge16, ego16, ego_id, ei1);

    #define GEMM(A, lda, W, bias, C, f32, rs, Mv, Nv, Kv, relu) \
        gemm_f16_wmma<<<dim3((Mv)/128, (Nv)/128), 256, 0, stream>>>(A, lda, W, bias, C, f32, rs, Mv, Nv, Kv, relu)

    // kv = edge_input @ Wkv + bkv                        (M x 512, K=160)
    GEMM(ei1, 160, Wkv16, bkv, kv16, 0, nullptr, M, 512, 160, 0);
    // k = K @ Wik + bik ; v = V @ Wiv + biv              (M x 256, K=256)
    GEMM(kv16,       512, Wik16, bik, k16,  0, nullptr, M, 256, 256, 0);
    GEMM(kv16 + 256, 512, Wiv16, biv, v16b, 0, nullptr, M, 256, 256, 0);
    // Q = ego @ Wq + bq ; q = Q @ Wiq + biq              (B rows)
    GEMM(ego16, 64, Wq16, bq, Q16, 0, nullptr, B_DIM, 256, 64, 0);
    GEMM(Q16, 256, Wiq16, biq, qf32, 1, nullptr, B_DIM, 256, 256, 0);

    // attention
    attn_kernel<<<B_DIM, 256, 0, stream>>>(qf32, k16, v16b, ctx16);

    // c_ego = ctx @ Wo + bo
    GEMM(ctx16, 256, Wo16, bo, cego16, 0, nullptr, B_DIM, 256, 256, 0);
    // h_ego_prime = relu([ego|c_ego] @ Wf1 + bf1) @ Wf2 + bf2 -> d_out
    build_h_kernel<<<(B_DIM * 320) / 256, 256, 0, stream>>>(ego16, cego16, hbuf);
    GEMM(hbuf, 320, Wf116, bf1, hmid16, 0, nullptr, B_DIM, 256, 320, 1);
    GEMM(hmid16, 256, Wf216, bf2, out_h, 1, nullptr, B_DIM, 256, 256, 0);

    // hidden = relu(edge_input2 @ Ws1 + bs1)             (M x 256, K=352)
    build_edge_input2_kernel<<<(M * 352) / 256, 256, 0, stream>>>(node16, edge16, cego16, ego_id, ei2);
    GEMM(ei2, 352, Ws116, bs1, hidden, 0, nullptr, M, 256, 352, 1);

    // score column + top-k + alpha
    score_kernel<<<M / 256, 256, 0, stream>>>(hidden, Ws2, bs2, scores);
    topk_kernel<<<B_DIM, 256, 0, stream>>>(scores, out_idx, out_alpha, alphab, idxb);

    // selected = (hidden[sel] @ Ws2[:,1:] + bs2[1:]) * alpha  -> d_out
    gather_kernel<<<(1280 * 256) / 256, 256, 0, stream>>>(hidden, idxb, gath16);
    GEMM(gath16, 256, Ws2c16, bs2 + 1, out_sel, 1, alphab, 1280, 1536, 256, 0);
    #undef GEMM
}
